// GraphAttentionLayer_65403761983861
// MI455X (gfx1250) — matrix operand order repaired and checked by reading, not compiled, not run on hardware
//
#include <hip/hip_runtime.h>

// ---------------------------------------------------------------------------
// Fused GAT layer for MI455X (gfx1250, wave32, WMMA bf16).
//   h = x@W ; e = lrelu(src_i + dst_j) ; alpha = softmax_j(e)
//   out = relu(LN(alpha @ h) * gamma + beta)
// Memory-bound (~0.5 GB HBM traffic): x read once, out written once,
// everything else lives in LDS / registers / L2.
// ---------------------------------------------------------------------------

typedef __attribute__((ext_vector_type(16))) __bf16 v16bf;
typedef __attribute__((ext_vector_type(8)))  __bf16 v8bf;
typedef __attribute__((ext_vector_type(8)))  float  v8f;
typedef __attribute__((ext_vector_type(4)))  float  v4f;

#define B_    32
#define T_    256
#define C_    64
#define FIN   128
#define FOUT  128
#define NEG_SLOPE 0.2f
#define LN_EPS    1e-5f

#define HT_STRIDE  (C_ + 8)     // 72 bf16 = 144 B; rows hit distinct bank groups
#define AL_STRIDE  (C_ + 8)     // 72 bf16
#define OUT_STRIDE (FOUT + 4)   // 132 f32 = 528 B (16B aligned rows)

__device__ __forceinline__ v16bf combine8(v8bf lo, v8bf hi) {
  v16bf r;
#pragma unroll
  for (int e = 0; e < 8; ++e) { r[e] = lo[e]; r[e + 8] = hi[e]; }
  return r;
}

// --- one-shot: Wt[fout][fin] = bf16(W[fin][fout])  (32 KB, stays in L2) ----
__global__ __launch_bounds__(256) void wt_cvt_kernel(
    const float* __restrict__ W, unsigned short* __restrict__ Wt_u16) {
  int i = blockIdx.x * 256 + threadIdx.x;          // 16384 elements
  int fin  = i >> 7;
  int fout = i & 127;
  ((__bf16*)Wt_u16)[fout * FIN + fin] = (__bf16)W[fin * FOUT + fout];
}

// --- main fused kernel: 1 workgroup per (b,t) tile, 128 threads = 4 waves --
__global__ __launch_bounds__(128) void gat_fused_kernel(
    const float* __restrict__ x,
    const unsigned short* __restrict__ Wt_u16,
    const float* __restrict__ attn_src,
    const float* __restrict__ attn_dst,
    const float* __restrict__ gamma,
    const float* __restrict__ beta,
    float* __restrict__ out) {
  const __bf16* Wt = (const __bf16*)Wt_u16;
  const int bt   = blockIdx.x;                     // 0..8191
  const int tid  = threadIdx.x;                    // 0..127
  const int wave = tid >> 5;                       // 0..3
  const int lane = tid & 31;
  const int half = lane >> 4;                      // 0/1 (WMMA lane halves)
  const int l16  = lane & 15;
  const int m0   = wave * 16;                      // wave's row strip

  __shared__ __bf16 sHt[FOUT * HT_STRIDE];         // h transposed, bf16 (GEMM2 B)
  __shared__ __bf16 sAl[C_ * AL_STRIDE];           // alpha bf16 (GEMM2 A)
  __shared__ float  sOut[C_ * OUT_STRIDE];         // GEMM2 result / softmax scratch
  __shared__ float  sSrc[C_], sDst[C_], sMu[C_], sRsig[C_];

  const v8f vzero = {0.f, 0.f, 0.f, 0.f, 0.f, 0.f, 0.f, 0.f};

  // ===== GEMM1: h(16x128 per wave) = x(16x128) @ W(128x128), bf16 WMMA =====
  v8f acc[8];
#pragma unroll
  for (int n = 0; n < 8; ++n) acc[n] = vzero;

  // A-fragment rows come straight from global fp32 (read exactly once).
  const float* xrow = x + ((size_t)bt * C_ + (m0 + l16)) * FIN;
#pragma unroll
  for (int kk = 0; kk < FIN; kk += 32) {
    // CDNA5 16-bit A layout: lane l16 holds row, chunks K=[kk+8h, +8) and
    // K=[kk+16+8h, +8). Load fp32, convert to bf16 in regs.
    const v4f* ap0 = (const v4f*)(xrow + kk + half * 8);
    const v4f* ap1 = (const v4f*)(xrow + kk + 16 + half * 8);
    v4f a0 = ap0[0], a1 = ap0[1], a2 = ap1[0], a3 = ap1[1];
    v16bf A;
#pragma unroll
    for (int e = 0; e < 4; ++e) {
      A[e]      = (__bf16)a0[e];
      A[e + 4]  = (__bf16)a1[e];
      A[e + 8]  = (__bf16)a2[e];
      A[e + 12] = (__bf16)a3[e];
    }
#pragma unroll
    for (int n = 0; n < 8; ++n) {
      // B layout: lane = column N, K chunks [kk+8*half,+8) and [kk+16+8*half,+8).
      const __bf16* bptr = Wt + (n * 16 + l16) * FIN + kk + half * 8;
      v16bf Bf = combine8(*(const v8bf*)bptr, *(const v8bf*)(bptr + 16));
      acc[n] = __builtin_amdgcn_wmma_f32_16x16x32_bf16(
          false, A, false, Bf, (short)0, acc[n], false, false);
    }
  }

  // Scatter h -> LDS, bf16, transposed [f][c] (== GEMM2 B-fragment layout).
#pragma unroll
  for (int n = 0; n < 8; ++n) {
#pragma unroll
    for (int r = 0; r < 8; ++r) {
      int m = m0 + r + half * 8;        // C/D f32 layout: VGPR r -> M = r + 8*half
      int f = n * 16 + l16;             // N = lane
      sHt[f * HT_STRIDE + m] = (__bf16)acc[n][r];
    }
  }
  __syncthreads();

  // ===== attention scores: src/dst = h . attn vectors (1 thread per node) ==
  if (tid < C_) {
    float s1 = 0.f, s2 = 0.f;
#pragma unroll 4
    for (int f = 0; f < FOUT; ++f) {
      float hv = (float)sHt[f * HT_STRIDE + tid];
      s1 = fmaf(hv, attn_src[f], s1);
      s2 = fmaf(hv, attn_dst[f], s2);
    }
    sSrc[tid] = s1;
    sDst[tid] = s2;
  }
  __syncthreads();

  // ===== leaky-relu + row softmax over 64 neighbors (1 thread per row) =====
  if (tid < C_) {
    float si = sSrc[tid];
    float mx = -3.0e38f;
#pragma unroll 4
    for (int j = 0; j < C_; ++j) {
      float e = si + sDst[j];
      e = (e >= 0.f) ? e : NEG_SLOPE * e;
      mx = fmaxf(mx, e);
      sOut[tid * OUT_STRIDE + j] = e;          // stash raw scores
    }
    float sum = 0.f;
#pragma unroll 4
    for (int j = 0; j < C_; ++j) {
      float p = __expf(sOut[tid * OUT_STRIDE + j] - mx);
      sOut[tid * OUT_STRIDE + j] = p;
      sum += p;
    }
    float rinv = 1.0f / sum;
#pragma unroll 4
    for (int j = 0; j < C_; ++j)
      sAl[tid * AL_STRIDE + j] = (__bf16)(sOut[tid * OUT_STRIDE + j] * rinv);
  }
  __syncthreads();

  // ===== GEMM2: out(16x128 per wave) = alpha(16x64) @ h(64x128) =============
  v8f acc2[8];
#pragma unroll
  for (int n = 0; n < 8; ++n) acc2[n] = vzero;
#pragma unroll
  for (int kk = 0; kk < C_; kk += 32) {
    const __bf16* arow = sAl + (m0 + l16) * AL_STRIDE;
    v16bf A = combine8(*(const v8bf*)(arow + kk + half * 8),
                       *(const v8bf*)(arow + kk + 16 + half * 8));
#pragma unroll
    for (int n = 0; n < 8; ++n) {
      const __bf16* bptr = sHt + (n * 16 + l16) * HT_STRIDE + kk + half * 8;
      v16bf Bf = combine8(*(const v8bf*)bptr, *(const v8bf*)(bptr + 16));
      acc2[n] = __builtin_amdgcn_wmma_f32_16x16x32_bf16(
          false, A, false, Bf, (short)0, acc2[n], false, false);
    }
  }

  // ===== LayerNorm + ReLU epilogue via LDS ==================================
#pragma unroll
  for (int n = 0; n < 8; ++n) {
#pragma unroll
    for (int r = 0; r < 8; ++r) {
      int m = m0 + r + half * 8;
      int f = n * 16 + l16;
      sOut[m * OUT_STRIDE + f] = acc2[n][r];
    }
  }
  __syncthreads();

  if (tid < C_) {
    float s = 0.f, s2 = 0.f;
#pragma unroll 4
    for (int f = 0; f < FOUT; ++f) {
      float v = sOut[tid * OUT_STRIDE + f];
      s += v;
      s2 = fmaf(v, v, s2);
    }
    float mu  = s * (1.0f / FOUT);
    float var = s2 * (1.0f / FOUT) - mu * mu;
    sMu[tid]   = mu;
    sRsig[tid] = rsqrtf(var + LN_EPS);
  }
  __syncthreads();

  // Coalesced float4 store: 8192 f32 per tile, 128 threads x 16 iters.
  float* obase = out + (size_t)bt * C_ * FOUT;
#pragma unroll
  for (int it = 0; it < 16; ++it) {
    int q   = tid + it * 128;          // float4 group id, 0..2047
    int row = q >> 5;                  // 32 float4 per row
    int col = (q & 31) * 4;
    float mu = sMu[row], rs = sRsig[row];
    v4f v  = *(const v4f*)(sOut + row * OUT_STRIDE + col);
    v4f g  = *(const v4f*)(gamma + col);
    v4f bb = *(const v4f*)(beta + col);
    v4f o;
#pragma unroll
    for (int e = 0; e < 4; ++e) {
      float t = (v[e] - mu) * rs * g[e] + bb[e];
      o[e] = t > 0.f ? t : 0.f;
    }
    *(v4f*)(obase + row * FOUT + col) = o;
  }
}

extern "C" void kernel_launch(void* const* d_in, const int* in_sizes, int n_in,
                              void* d_out, int out_size, void* d_ws, size_t ws_size,
                              hipStream_t stream) {
  (void)in_sizes; (void)n_in; (void)out_size; (void)ws_size;
  const float* x        = (const float*)d_in[0];  // (32,256,64,128)
  const float* W        = (const float*)d_in[1];  // (128,128)
  const float* attn_src = (const float*)d_in[2];  // (128,)
  const float* attn_dst = (const float*)d_in[3];  // (128,)
  const float* gamma    = (const float*)d_in[4];  // (128,)
  const float* beta     = (const float*)d_in[5];  // (128,)
  float* out            = (float*)d_out;          // (32,256,64,128)

  unsigned short* Wt = (unsigned short*)d_ws;     // 32 KB bf16, transposed W

  wt_cvt_kernel<<<64, 256, 0, stream>>>(W, Wt);
  gat_fused_kernel<<<B_ * T_, 128, 0, stream>>>(x, Wt, attn_src, attn_dst,
                                                gamma, beta, out);
}
